// VentilatorNet_6674379178048
// MI455X (gfx1250) — matrix operand order repaired and checked by reading, not compiled, not run on hardware
//
#include <hip/hip_runtime.h>
#include <hip/hip_bf16.h>

// ---------------------------------------------------------------------------
// VentilatorNet: 4x BiLSTM (H=1024,512,256,128) + FC(256->50)+SELU+FC(50->1)
// CDNA5 (gfx1250, wave32) implementation using v_wmma_f32_16x16x32_f16 and
// async global->LDS staging (ASYNCcnt path).
// ---------------------------------------------------------------------------

typedef _Float16 v16h __attribute__((ext_vector_type(16)));
typedef _Float16 v8h  __attribute__((ext_vector_type(8)));
typedef float    v8f  __attribute__((ext_vector_type(8)));

// D = A(16x32) * B(32x16) + C, f32 accumulate
__device__ __forceinline__ v8f wmma_f16(v16h a, v16h b, v8f c) {
    return __builtin_amdgcn_wmma_f32_16x16x32_f16(
        /*neg_a=*/false, a, /*neg_b=*/false, b,
        /*c_mod=*/(short)0, c, /*reuse_a=*/false, /*reuse_b=*/false);
}

// A fragment: 16x32 (MxK), row-major source with leading dim `ld`.
// Lane l: row m = l&15; kh = l>>4 selects K halves per CDNA5 ISA layout:
//   VGPR0-3 : K = k0 + kh*8 + {0..7}
//   VGPR4-7 : K = k0 + 16 + kh*8 + {0..7}
__device__ __forceinline__ v16h load_a_frag(const _Float16* base, int ld, int k0, int lane) {
    const int m  = lane & 15;
    const int kh = lane >> 4;
    const _Float16* p = base + (size_t)m * ld + k0 + kh * 8;
    v8h lo = *(const v8h*)(p);
    v8h hi = *(const v8h*)(p + 16);
    return __builtin_shufflevector(lo, hi, 0,1,2,3,4,5,6,7,8,9,10,11,12,13,14,15);
}

// B fragment: 32x16 (KxN) where B[k][n] = W[n0+n][k] (weight row-major [N][K]).
// Lane l: column n = l&15, K = k0 + (l>>4)*8 + {0..7} and k0 + 16 + (l>>4)*8 + {0..7}
// (same K order as the A fragment), read from a weight row, no transpose needed.
__device__ __forceinline__ v16h load_b_frag(const _Float16* W, int ldk, int n0, int k0, int lane) {
    const int n  = lane & 15;
    const int kh = lane >> 4;
    const _Float16* p = W + (size_t)(n0 + n) * ldk + k0 + kh * 8;
    v8h lo = *(const v8h*)(p);
    v8h hi = *(const v8h*)(p + 16);
    return __builtin_shufflevector(lo, hi, 0,1,2,3,4,5,6,7,8,9,10,11,12,13,14,15);
}

__device__ __forceinline__ float sigm_(float x)  { return 1.f / (1.f + __expf(-x)); }
__device__ __forceinline__ float tanh_(float x)  { return 2.f / (1.f + __expf(-2.f * x)) - 1.f; }

// ---------------------------------------------------------------------------
// f32 -> f16 cast with zero padding (rows and cols). Used for activations and
// weight conversion (pad K of layer-0 Wih 50->64, pad fc1_w rows 50->64).
// ---------------------------------------------------------------------------
__global__ void k_cast_pad(const float* __restrict__ src, _Float16* __restrict__ dst,
                           int src_rows, int src_cols, int dst_rows, int dst_cols) {
    int idx = blockIdx.x * blockDim.x + threadIdx.x;
    int total = dst_rows * dst_cols;
    if (idx >= total) return;
    int r = idx / dst_cols;
    int c = idx - r * dst_cols;
    float v = (r < src_rows && c < src_cols) ? src[(size_t)r * src_cols + c] : 0.f;
    dst[idx] = (_Float16)v;
}

// ---------------------------------------------------------------------------
// gx = A[M,K](f16) @ W[N,K]^T + bih + bhh  -> gx[M,N] (f16), both directions
// in one launch (blockIdx.z). Block = 256 threads = 8 waves; each block does a
// 16(M) x 128(N) tile. The A row-block is staged into LDS with CDNA5 async
// global->LDS copies (no VGPR round-trip, tracked by ASYNCcnt) and then shared
// by all 8 waves.
// ---------------------------------------------------------------------------
#define GX_MAXK 2048
__global__ void k_gx_gemm(const _Float16* __restrict__ A, int K,
                          const _Float16* __restrict__ Wf, const _Float16* __restrict__ Wb,
                          const float* __restrict__ bihF, const float* __restrict__ bhhF,
                          const float* __restrict__ bihB, const float* __restrict__ bhhB,
                          _Float16* __restrict__ gxF, _Float16* __restrict__ gxB, int N) {
    __shared__ __align__(32) _Float16 sA[16 * GX_MAXK];   // 64 KB max

    const int tid  = threadIdx.x;
    const int lane = tid & 31;
    const int wave = tid >> 5;
    const int row0 = blockIdx.y * 16;
    const int n0   = blockIdx.x * 128 + wave * 16;
    const int dir  = blockIdx.z;

    const _Float16* W   = dir ? Wb   : Wf;
    const float*    bih = dir ? bihB : bihF;
    const float*    bhh = dir ? bhhB : bhhF;
    _Float16*       gx  = dir ? gxB  : gxF;

    // Stage A tile [16][K] -> LDS via async 16B copies. For a generic pointer
    // into the LDS aperture, addr[31:0] is the LDS byte address (ISA 10.2),
    // which is what VDST of global_load_async_to_lds_b128 expects.
    {
        const _Float16* src = A + (size_t)row0 * K;
        const int nvec = 2 * K;  // 16*K f16 = 2*K chunks of 16 bytes
        for (int i = tid; i < nvec; i += blockDim.x) {
            unsigned lds_addr = (unsigned)(uintptr_t)(sA + (size_t)i * 8);
            unsigned long long gaddr = (unsigned long long)(uintptr_t)(src + (size_t)i * 8);
            asm volatile("global_load_async_to_lds_b128 %0, %1, off"
                         :: "v"(lds_addr), "v"(gaddr) : "memory");
        }
        asm volatile("s_wait_asynccnt 0x0" ::: "memory");
    }
    __syncthreads();

    v8f acc = {};
    for (int k0 = 0; k0 < K; k0 += 32) {
        v16h a = load_a_frag(sA, K, k0, lane);
        v16h b = load_b_frag(W, K, n0, k0, lane);
        if (k0 + 32 < K)  // warm L2->L0 path for the next weight chunk
            __builtin_prefetch(W + (size_t)(n0 + (lane & 15)) * K + k0 + 32, 0, 1);
        acc = wmma_f16(a, b, acc);
    }

    const int n  = n0 + (lane & 15);
    const int kh = lane >> 4;
    const float bias = bih[n] + bhh[n];
    #pragma unroll
    for (int v = 0; v < 8; ++v) {
        int m = v + 8 * kh;
        gx[(size_t)(row0 + m) * N + n] = (_Float16)(acc[v] + bias);
    }
}

// ---------------------------------------------------------------------------
// Persistent bidirectional LSTM recurrence. Grid = (B/16, 2 directions).
// Each workgroup owns 16 batch rows; h state ping-pongs in LDS (f16), c state
// lives in registers (templated H -> fully unrolled tile loop). 8 waves each
// own H/8 hidden units and compute all four gate blocks for their slice, so
// the cell update is wave-local. One workgroup barrier per time step.
// All weights stay L2-resident (192 MB), so per-step Whh re-reads never touch
// HBM; the serial chain is bounded by L2 bandwidth + WMMA issue, not 23.3 TB/s.
// ---------------------------------------------------------------------------
template <int H>
__global__ void k_bilstm(const _Float16* __restrict__ gxF, const _Float16* __restrict__ gxB,
                         const _Float16* __restrict__ WhhF, const _Float16* __restrict__ WhhB,
                         _Float16* __restrict__ y, int T) {
    constexpr int N4 = 4 * H;
    constexpr int HW = H / 8;    // hidden units per wave
    constexpr int NT = HW / 16;  // 16-wide tiles per wave

    __shared__ __align__(32) _Float16 sH0[16 * H];
    __shared__ __align__(32) _Float16 sH1[16 * H];

    const int tid  = threadIdx.x;
    const int lane = tid & 31;
    const int wave = tid >> 5;
    const int b0   = blockIdx.x * 16;
    const int dir  = blockIdx.y;

    const _Float16* gx  = dir ? gxB  : gxF;
    const _Float16* Whh = dir ? WhhB : WhhF;

    for (int i = tid; i < 16 * H; i += blockDim.x) sH0[i] = (_Float16)0;
    __syncthreads();

    const int nl = lane & 15;
    const int kh = lane >> 4;
    const size_t rowstride = (size_t)T * N4;

    float creg[NT * 8] = {};  // cell state, register-resident

    for (int step = 0; step < T; ++step) {
        const int tt = dir ? (T - 1 - step) : step;
        const _Float16* hcur = (step & 1) ? sH1 : sH0;
        _Float16*       hnxt = (step & 1) ? sH0 : sH1;
        const size_t gbase = ((size_t)b0 * T + tt) * N4;

        #pragma unroll
        for (int ht = 0; ht < NT; ++ht) {
            const int j0 = wave * HW + ht * 16;  // hidden-unit tile base
            v8f ai = {}, af = {}, ag = {}, ao = {};
            for (int k0 = 0; k0 < H; k0 += 32) {
                v16h a  = load_a_frag(hcur, H, k0, lane);
                v16h bi = load_b_frag(Whh, H, 0 * H + j0, k0, lane);
                v16h bf = load_b_frag(Whh, H, 1 * H + j0, k0, lane);
                v16h bg = load_b_frag(Whh, H, 2 * H + j0, k0, lane);
                v16h bo = load_b_frag(Whh, H, 3 * H + j0, k0, lane);
                ai = wmma_f16(a, bi, ai);
                af = wmma_f16(a, bf, af);
                ag = wmma_f16(a, bg, ag);
                ao = wmma_f16(a, bo, ao);
            }
            const int j = j0 + nl;
            #pragma unroll
            for (int v = 0; v < 8; ++v) {
                const int m = v + 8 * kh;
                const size_t g = gbase + (size_t)m * rowstride;
                float gi = ai[v] + (float)gx[g + 0 * H + j];
                float gf = af[v] + (float)gx[g + 1 * H + j];
                float gg = ag[v] + (float)gx[g + 2 * H + j];
                float go = ao[v] + (float)gx[g + 3 * H + j];
                float i_ = sigm_(gi);
                float f_ = sigm_(gf);   // forget bias (+1) already in gx via bih
                float g_ = tanh_(gg);
                float o_ = sigm_(go);
                float c  = f_ * creg[ht * 8 + v] + i_ * g_;
                float h  = o_ * tanh_(c);
                creg[ht * 8 + v] = c;
                hnxt[m * H + j]  = (_Float16)h;
                y[((size_t)(b0 + m) * T + tt) * (2 * H) + dir * H + j] = (_Float16)h;
            }
        }
        __syncthreads();  // h(t) fully written / h(t-1) fully consumed
    }
}

// ---------------------------------------------------------------------------
// Head: out = fc2( selu( A[M,256] @ fc1_w^T + fc1_b ) ). fc1_w padded to
// [64][256] (rows 50..63 zero). Block = 128 threads = 4 waves = 64 cols.
// ---------------------------------------------------------------------------
__global__ void k_head(const _Float16* __restrict__ A, const _Float16* __restrict__ W1,
                       const float* __restrict__ b1, const float* __restrict__ w2,
                       const float* __restrict__ b2, float* __restrict__ out) {
    __shared__ __align__(16) float sS[16 * 64];
    const int tid  = threadIdx.x;
    const int lane = tid & 31;
    const int wave = tid >> 5;       // 0..3
    const int row0 = blockIdx.x * 16;
    const int K = 256;
    const int n0 = wave * 16;

    v8f acc = {};
    for (int k0 = 0; k0 < K; k0 += 32) {
        v16h a = load_a_frag(A + (size_t)row0 * K, K, k0, lane);
        v16h b = load_b_frag(W1, K, n0, k0, lane);
        acc = wmma_f16(a, b, acc);
    }

    const int n  = n0 + (lane & 15);
    const int kh = lane >> 4;
    const float bias = (n < 50) ? b1[n] : 0.f;
    const float SC = 1.0507009873554804934f;
    const float AL = 1.6732632423543772848f;
    #pragma unroll
    for (int v = 0; v < 8; ++v) {
        int m = v + 8 * kh;
        float x = acc[v] + bias;
        sS[m * 64 + n] = (x > 0.f) ? SC * x : SC * AL * (__expf(x) - 1.f);
    }
    __syncthreads();
    if (tid < 16) {
        float sum = b2[0];
        for (int k = 0; k < 50; ++k) sum += sS[tid * 64 + k] * w2[k];
        out[row0 + tid] = sum;
    }
}

// ---------------------------------------------------------------------------
// Host orchestration. Input order (flattened setup_inputs dict):
//   [0] x[256,80,50]
//   layer l (l=0..3), base=1+8l:
//     +0 WihF[4H,din] +1 WhhF[4H,H] +2 bihF[4H] +3 bhhF[4H]
//     +4 WihB         +5 WhhB      +6 bihB     +7 bhhB
//   [33] fc1_w[50,256] [34] fc1_b[50] [35] fc2_w[1,50] [36] fc2_b[1]
// ---------------------------------------------------------------------------
extern "C" void kernel_launch(void* const* d_in, const int* in_sizes, int n_in,
                              void* d_out, int out_size, void* d_ws, size_t ws_size,
                              hipStream_t stream) {
    (void)in_sizes; (void)n_in; (void)out_size; (void)ws_size;

    const int B = 256, T = 80;
    const int M = B * T;                               // 20480 rows
    const int Hs[4]    = {1024, 512, 256, 128};
    const int dins[4]  = {50, 2048, 1024, 512};
    const int dpads[4] = {64, 2048, 1024, 512};        // K padded to mult of 32

    // Workspace layout (~537 MB total)
    char* ws = (char*)d_ws;
    size_t off = 0;
    auto bump = [&](size_t bytes) -> char* {
        char* p = ws + off;
        off += (bytes + 255) & ~(size_t)255;
        return p;
    };
    _Float16* bufA   = (_Float16*)bump((size_t)M * 2048 * sizeof(_Float16)); // 84 MB
    _Float16* bufB   = (_Float16*)bump((size_t)M * 2048 * sizeof(_Float16)); // 84 MB
    _Float16* wih0   = (_Float16*)bump((size_t)8 << 20);                     // 8 MB
    _Float16* wih1   = (_Float16*)bump((size_t)8 << 20);
    _Float16* whh0   = (_Float16*)bump((size_t)8 << 20);
    _Float16* whh1   = (_Float16*)bump((size_t)8 << 20);
    _Float16* w1pad  = (_Float16*)bump((size_t)64 * 256 * sizeof(_Float16));
    _Float16* gx0    = (_Float16*)bump((size_t)M * 4096 * sizeof(_Float16)); // 168 MB
    _Float16* gx1    = (_Float16*)bump((size_t)M * 4096 * sizeof(_Float16)); // 168 MB

    auto cast = [&](const float* s, _Float16* d, int sr, int sc, int dr, int dc) {
        int total = dr * dc;
        k_cast_pad<<<(total + 255) / 256, 256, 0, stream>>>(s, d, sr, sc, dr, dc);
    };

    // Layer-0 input: [M,50] f32 -> [M,64] f16 (zero padded)
    cast((const float*)d_in[0], bufA, M, 50, M, 64);
    // fc1 weight: [50,256] -> [64,256] f16 (zero rows 50..63)
    cast((const float*)d_in[33], w1pad, 50, 256, 64, 256);

    _Float16* cur = bufA;
    _Float16* nxt = bufB;

    for (int l = 0; l < 4; ++l) {
        const int H = Hs[l], din = dins[l], K = dpads[l], N = 4 * H;
        const int base = 1 + l * 8;
        const float* WihF = (const float*)d_in[base + 0];
        const float* WhhF = (const float*)d_in[base + 1];
        const float* bihF = (const float*)d_in[base + 2];
        const float* bhhF = (const float*)d_in[base + 3];
        const float* WihB = (const float*)d_in[base + 4];
        const float* WhhB = (const float*)d_in[base + 5];
        const float* bihB = (const float*)d_in[base + 6];
        const float* bhhB = (const float*)d_in[base + 7];

        // Convert weights to f16 (kept in PyTorch [rows][K] layout = B-operand ready)
        cast(WihF, wih0, N, din, N, K);
        cast(WihB, wih1, N, din, N, K);
        cast(WhhF, whh0, N, H, N, H);
        cast(WhhB, whh1, N, H, N, H);

        // gx = x @ Wih^T + bih + bhh, both directions in one launch
        dim3 g1(N / 128, M / 16, 2);
        k_gx_gemm<<<g1, 256, 0, stream>>>(cur, K, wih0, wih1,
                                          bihF, bhhF, bihB, bhhB, gx0, gx1, N);

        // recurrence, both directions in one launch
        dim3 g2(B / 16, 2);
        switch (H) {
            case 1024: k_bilstm<1024><<<g2, 256, 0, stream>>>(gx0, gx1, whh0, whh1, nxt, T); break;
            case 512:  k_bilstm<512> <<<g2, 256, 0, stream>>>(gx0, gx1, whh0, whh1, nxt, T); break;
            case 256:  k_bilstm<256> <<<g2, 256, 0, stream>>>(gx0, gx1, whh0, whh1, nxt, T); break;
            default:   k_bilstm<128> <<<g2, 256, 0, stream>>>(gx0, gx1, whh0, whh1, nxt, T); break;
        }

        _Float16* tmp = cur; cur = nxt; nxt = tmp;  // layer output width 2H == next K
    }

    // Head: cur = [M,256] f16
    k_head<<<M / 16, 128, 0, stream>>>(cur, w1pad,
                                       (const float*)d_in[34], (const float*)d_in[35],
                                       (const float*)d_in[36], (float*)d_out);
}
